// GemsNetT_67783173865706
// MI455X (gfx1250) — matrix operand order repaired and checked by reading, not compiled, not run on hardware
//
#include <hip/hip_runtime.h>
#include <hip/hip_bf16.h>
#include <math.h>

typedef __hip_bfloat16 bf16;
typedef __attribute__((ext_vector_type(16))) __bf16 v16bf;
typedef __attribute__((ext_vector_type(8)))  float  v8f;
typedef __attribute__((ext_vector_type(4)))  unsigned int v4u;
typedef __attribute__((ext_vector_type(8)))  int v8i;
typedef __attribute__((ext_vector_type(4)))  int v4i;

#define INV_SQRT2F 0.70710678118654752f
#define ACT_SCALE  (1.0f/0.6f)

#if defined(__AMDGCN__) && __has_builtin(__builtin_amdgcn_tensor_load_to_lds)
#define USE_TDM 1
#else
#define USE_TDM 0
#endif

__device__ __forceinline__ float scaled_silu(float x) {
  return x * (1.0f / (1.0f + __expf(-x))) * ACT_SCALE;
}

struct uint4x2 { uint4 a, b; };
union FragU { uint4x2 u; v16bf v; };

#if USE_TDM
// Issue one 2D TDM descriptor: copy tile [tile1 rows x tile0 elems] of 2-byte
// data from global (row stride = stride0 elems, row length clamped to td0 for
// OOB zero-fill) into LDS at byte offset lds_addr, packed row-major.
// D# layout per CDNA5 ISA ch.8 (group0 128b, group1 256b; groups2/3 zero).
// This toolchain exposes the 6-arg builtin (g0, g1, g2, g3, g4, cpol).
__device__ __forceinline__ void tdm_load_2d(unsigned int lds_addr, const void* gptr,
                                            unsigned int td0, unsigned int td1,
                                            unsigned int tile0, unsigned int tile1,
                                            unsigned int stride0) {
  unsigned long long ga = (unsigned long long)gptr;
  v4u g0;
  g0[0] = 1u;                                            // count=1 (valid), user mode
  g0[1] = lds_addr;                                      // lds_addr [63:32]
  g0[2] = (unsigned int)(ga & 0xFFFFFFFFu);              // global_addr[31:0]
  g0[3] = (unsigned int)((ga >> 32) & 0x01FFFFFFu)       // global_addr[56:32]
        | (2u << 30);                                    // type=2 ("image")
  v8i g1;
  g1[0] = (int)(1u << 16);                               // data_size=1 -> 2 bytes
  g1[1] = (int)((td0 & 0xFFFFu) << 16);                  // tensor_dim0[15:0] @ b48
  g1[2] = (int)((td0 >> 16) | ((td1 & 0xFFFFu) << 16));  // dim0 hi | dim1 lo
  g1[3] = (int)((td1 >> 16) | (tile0 << 16));            // dim1 hi | tile_dim0
  g1[4] = (int)(tile1);                                  // tile_dim1 (tile_dim2=0)
  g1[5] = (int)(stride0);                                // tensor_dim0_stride[31:0]
  g1[6] = 0;                                             // stride0 hi | stride1 lo
  g1[7] = 0;                                             // stride1 hi
  v4i zz4 = {0, 0, 0, 0};
  v8i zz8 = {0, 0, 0, 0, 0, 0, 0, 0};
  __builtin_amdgcn_tensor_load_to_lds(g0, g1, zz4, zz4, zz8, 0);
}
#endif

// ---------------------------------------------------------------------------
// WMMA GEMM: Y[M,N] = epilogue( X[M,K](bf16,row-major,ldx) @ Wt^T ), Wt is
// the transposed weight stored [N,K] bf16 row-major. NT = N/16 output tiles.
// mode: 0 none, 1 act, 2 (Res+act(acc))*inv_sqrt2, 3 act(acc)*Gate, 4 acc*Gate
// ---------------------------------------------------------------------------
template<int NT>
__global__ __launch_bounds__(256) void wmma_gemm_kernel(
    const bf16* __restrict__ X, int ldx,
    const bf16* __restrict__ Wt,
    float* __restrict__ Yf, bf16* __restrict__ Yb, int ldy,
    const float* __restrict__ Res, const float* __restrict__ Gate,
    int K, int mode)
{
  __shared__ bf16 Xs[128 * 32];
  __shared__ bf16 Ws[NT * 16 * 32];
  const int tid  = threadIdx.x;
  const int wave = tid >> 5;
  const int lane = tid & 31;
  const int hlf  = lane >> 4;
  const int l16  = lane & 15;
  const int m0   = blockIdx.x * 128;

  v8f acc[NT] = {};

  const int kchunks = (K + 31) >> 5;
#if USE_TDM
  const unsigned int xs_lds = (unsigned int)(size_t)(void*)Xs;
  const unsigned int ws_lds = (unsigned int)(size_t)(void*)Ws;
#else
  const uint32_t* __restrict__ Xg = (const uint32_t*)X;
  const uint32_t* __restrict__ Wg = (const uint32_t*)Wt;
  uint32_t* Xls = (uint32_t*)Xs;
  uint32_t* Wls = (uint32_t*)Ws;
  const int ldx2 = ldx >> 1;
  const int ldw2 = K >> 1;
#endif

  for (int kc = 0; kc < kchunks; ++kc) {
    const int kbase = kc << 5;
    __syncthreads();
#if USE_TDM
    // Tensor Data Mover staging: wave 0 pulls the X tile, wave 1 the W tile.
    // tensor_dim0 = K-kbase gives hardware zero-fill past the K edge.
    if (wave == 0) {
      tdm_load_2d(xs_lds, X + (size_t)m0 * ldx + kbase,
                  (unsigned int)(K - kbase), 128u, 32u, 128u, (unsigned int)ldx);
      __builtin_amdgcn_s_wait_tensorcnt(0);
    } else if (wave == 1) {
      tdm_load_2d(ws_lds, Wt + kbase,
                  (unsigned int)(K - kbase), (unsigned int)(NT * 16), 32u,
                  (unsigned int)(NT * 16), (unsigned int)K);
      __builtin_amdgcn_s_wait_tensorcnt(0);
    }
#else
    // cooperative staging fallback
    #pragma unroll
    for (int it = 0; it < 8; ++it) {
      int idx = tid + it * 256;
      int row = idx >> 4;
      int dw  = idx & 15;
      int k   = kbase + (dw << 1);
      uint32_t v = 0u;
      if (k < K) v = Xg[(size_t)(m0 + row) * ldx2 + (k >> 1)];
      Xls[(row << 4) + dw] = v;
    }
    for (int idx = tid; idx < NT * 256; idx += 256) {
      int row = idx >> 4;
      int dw  = idx & 15;
      int k   = kbase + (dw << 1);
      uint32_t v = 0u;
      if (k < K) v = Wg[(size_t)row * ldw2 + (k >> 1)];
      Wls[(row << 4) + dw] = v;
    }
#endif
    if (kc + 1 < kchunks) {
      __builtin_prefetch(&X[(size_t)(m0 + (tid >> 1)) * ldx + kbase + 32]);
    }
    __syncthreads();

    // A fragment: row = wave*16 + l16; VGPR0-3 k=hlf*8..+7, VGPR4-7 k=16+hlf*8..+7
    FragU af;
    const uint4* Xv = (const uint4*)(Xs + (((wave << 4) + l16) << 5));
    af.u.a = Xv[hlf];
    af.u.b = Xv[2 + hlf];
    #pragma unroll
    for (int nt = 0; nt < NT; ++nt) {
      // B fragment: col n = l16 of tile nt; VGPR0-3 k=hlf*8..+7, VGPR4-7 k=16+hlf*8..+7
      FragU bfr;
      const uint4* Wv = (const uint4*)(Ws + (((nt << 4) + l16) << 5));
      bfr.u.a = Wv[hlf];
      bfr.u.b = Wv[2 + hlf];
      acc[nt] = __builtin_amdgcn_wmma_f32_16x16x32_bf16(
          false, af.v, false, bfr.v, (short)0, acc[nt], false, false);
    }
  }

  // epilogue: C/D layout -> lane col = l16, row = hlf*8 + j
  #pragma unroll
  for (int nt = 0; nt < NT; ++nt) {
    #pragma unroll
    for (int j = 0; j < 8; ++j) {
      int row = m0 + (wave << 4) + (hlf << 3) + j;
      int col = (nt << 4) + l16;
      size_t off = (size_t)row * ldy + col;
      float v = acc[nt][j];
      if (mode == 1)      v = scaled_silu(v);
      else if (mode == 2) v = (Res[off] + scaled_silu(v)) * INV_SQRT2F;
      else if (mode == 3) v = scaled_silu(v) * Gate[off];
      else if (mode == 4) v = v * Gate[off];
      if (Yf) Yf[off] = v;
      if (Yb) Yb[off] = __float2bfloat16(v);
    }
  }
}

// ---------------------------------------------------------------------------
// Support kernels
// ---------------------------------------------------------------------------
__global__ void wprep_kernel(const float* __restrict__ src, bf16* __restrict__ dst,
                             int K, int N) {
  int i = blockIdx.x * 256 + threadIdx.x;
  if (i >= K * N) return;
  int n = i / K, k = i % K;                 // dst [N,K] <- src [K,N]
  dst[i] = __float2bfloat16(src[(size_t)k * N + n]);
}

__global__ void edge_vec_kernel(const float* __restrict__ cell, const float* __restrict__ e_ij,
                                const int* __restrict__ batch_edge,
                                float* __restrict__ V, float* __restrict__ D, int E) {
  int e = blockIdx.x * 256 + threadIdx.x;
  if (e >= E) return;
  const float* c = cell + batch_edge[e] * 9;
  float f0 = e_ij[e*3], f1 = e_ij[e*3+1], f2 = e_ij[e*3+2];
  float v0 = c[0]*f0 + c[1]*f1 + c[2]*f2;
  float v1 = c[3]*f0 + c[4]*f1 + c[5]*f2;
  float v2 = c[6]*f0 + c[7]*f1 + c[8]*f2;
  float d = sqrtf(v0*v0 + v1*v1 + v2*v2);
  D[e] = d;
  float inv = -1.0f / d;
  V[e*3] = v0*inv; V[e*3+1] = v1*inv; V[e*3+2] = v2*inv;
}

__global__ void rbf_kernel(const float* __restrict__ D, float* __restrict__ rf,
                           bf16* __restrict__ rb, int E) {
  int i = blockIdx.x * 256 + threadIdx.x;
  if (i >= E * 128) return;
  int e = i >> 7, r = i & 127;
  float d = D[e] * (1.0f / 6.0f);
  float env = 0.0f;
  if (d < 1.0f) {
    float d2 = d*d, d4 = d2*d2, d5 = d4*d, d6 = d5*d, d7 = d6*d;
    env = 1.0f - 21.0f*d5 + 35.0f*d6 - 15.0f*d7;
  }
  float t = d - (float)r * (1.0f / 127.0f);
  float val = env * __expf(-8064.5f * t * t);   // -0.5*(127^2)
  rf[i] = val;
  rb[i] = __float2bfloat16(val);
}

__global__ void sbf_kernel(const float* __restrict__ V, const int* __restrict__ ba,
                           const int* __restrict__ ca, float* __restrict__ sbf, int T) {
  int t = blockIdx.x * 256 + threadIdx.x;
  if (t >= T) return;
  int a = ca[t], b = ba[t];
  float c = V[a*3]*V[b*3] + V[a*3+1]*V[b*3+1] + V[a*3+2]*V[b*3+2];
  c = fminf(1.0f, fmaxf(-1.0f, c));
  float out[7];
  out[0] = 1.0f; out[1] = c;
  float Pm1 = 1.0f, P = c;
  #pragma unroll
  for (int l = 1; l < 6; ++l) {
    float Pn = ((float)(2*l+1) * c * P - (float)l * Pm1) / (float)(l+1);
    Pm1 = P; P = Pn; out[l+1] = Pn;
  }
  const float pref[7] = {0.28209479f, 0.48860251f, 0.63078313f, 0.74635267f,
                         0.84628438f, 0.93560258f, 1.01710723f};
  #pragma unroll
  for (int l = 0; l < 7; ++l) sbf[(size_t)t*7 + l] = out[l] * pref[l];
}

__global__ void embed_kernel(const float* __restrict__ emb, const int* __restrict__ z,
                             float* __restrict__ hf, bf16* __restrict__ hb, int NA) {
  int i = blockIdx.x * 256 + threadIdx.x;
  if (i >= NA * 128) return;
  int n = i >> 7, c = i & 127;
  float v = emb[(size_t)z[n]*128 + c];
  hf[i] = v; hb[i] = __float2bfloat16(v);
}

__global__ void concat_kernel(const bf16* __restrict__ hA, const int* __restrict__ idx_s,
                              const int* __restrict__ idx_t, const bf16* __restrict__ edgeC,
                              bf16* __restrict__ out, int E) {
  int i = blockIdx.x * 256 + threadIdx.x;
  if (i >= E * 384) return;
  int e = i / 384, c = i % 384;
  bf16 v;
  if (c < 128)      v = hA[(size_t)idx_s[e]*128 + c];
  else if (c < 256) v = hA[(size_t)idx_t[e]*128 + (c - 128)];
  else              v = edgeC[(size_t)e*128 + (c - 256)];
  out[i] = v;
}

__global__ void ew_mul_tobf(const float* __restrict__ a, const float* __restrict__ b,
                            bf16* __restrict__ o, int n) {
  int i = blockIdx.x * 256 + threadIdx.x;
  if (i < n) o[i] = __float2bfloat16(a[i] * b[i]);
}

__global__ void ew_addscale(const float* __restrict__ a, const float* __restrict__ b,
                            float* __restrict__ of, bf16* __restrict__ ob, int n) {
  int i = blockIdx.x * 256 + threadIdx.x;
  if (i >= n) return;
  float v = (a[i] + b[i]) * INV_SQRT2F;
  of[i] = v;
  if (ob) ob[i] = __float2bfloat16(v);
}

__global__ void ew_combine(const float* __restrict__ xca, const float* __restrict__ upA,
                           const float* __restrict__ upB, const int* __restrict__ sw,
                           float* __restrict__ of, bf16* __restrict__ ob, int E) {
  int i = blockIdx.x * 256 + threadIdx.x;
  if (i >= E * 128) return;
  int e = i >> 7, c = i & 127;
  float x3 = (upA[i] + upB[(size_t)sw[e]*128 + c]) * INV_SQRT2F;
  float v = (xca[i] + x3) * INV_SQRT2F;
  of[i] = v; ob[i] = __float2bfloat16(v);
}

__global__ void f2b_kernel(const float* __restrict__ a, bf16* __restrict__ o, int n) {
  int i = blockIdx.x * 256 + threadIdx.x;
  if (i < n) o[i] = __float2bfloat16(a[i]);
}

// triplet aggregation: acc[e][s][d] = sum_{i != j} sbf[t(n,i,j)][s] * xba[n*16+i][d]
__global__ void trip_acc_kernel(const float* __restrict__ sbf, const float* __restrict__ xba,
                                float* __restrict__ acc, int E) {
  int i = blockIdx.x * 256 + threadIdx.x;
  if (i >= E * 32) return;
  int e = i >> 5, d = i & 31;
  int node = e >> 4, j = e & 15;
  float a[7] = {0,0,0,0,0,0,0};
  for (int ii = 0; ii < 16; ++ii) {
    if (ii == j) continue;
    int t = node * 240 + ii * 15 + (j < ii ? j : j - 1);
    float xv = xba[(size_t)(node * 16 + ii) * 32 + d];
    const float* sp = sbf + (size_t)t * 7;
    #pragma unroll
    for (int s = 0; s < 7; ++s) a[s] += sp[s] * xv;
  }
  float* ap = acc + (size_t)e * 224 + d;
  #pragma unroll
  for (int s = 0; s < 7; ++s) ap[s * 32] = a[s];
}

// xq[e,d,t] = sum_s rbfW1[e,s,d] * acc[e,s,t]  (d<16, t<32) -> bf16 [E,512]
__global__ __launch_bounds__(256) void xq_kernel(const float* __restrict__ rbfW1,
                                                 const float* __restrict__ acc,
                                                 bf16* __restrict__ xq) {
  __shared__ float rs[112];
  __shared__ float as[224];
  int e = blockIdx.x, tid = threadIdx.x;
  if (tid < 112) rs[tid] = rbfW1[(size_t)e * 112 + tid];
  if (tid < 224) as[tid] = acc[(size_t)e * 224 + tid];
  __syncthreads();
  #pragma unroll
  for (int q = 0; q < 2; ++q) {
    int p = tid + q * 256;
    int d = p >> 5, t = p & 31;
    float sum = 0.0f;
    #pragma unroll
    for (int s = 0; s < 7; ++s) sum += rs[s * 16 + d] * as[s * 32 + t];
    xq[(size_t)e * 512 + p] = __float2bfloat16(sum);
  }
}

__global__ void scatter_atom_kernel(const float* __restrict__ xm, const float* __restrict__ gate,
                                    const int* __restrict__ idx_t, float* __restrict__ a, int E) {
  int i = blockIdx.x * 256 + threadIdx.x;
  if (i >= E * 128) return;
  int e = i >> 7, c = i & 127;
  atomicAdd(&a[(size_t)idx_t[e]*128 + c], xm[i] * gate[i]);
}

__global__ void outF_kernel(const float* __restrict__ xF, const float* __restrict__ W,
                            float* __restrict__ F_st, int E) {
  int e = blockIdx.x * 256 + threadIdx.x;
  if (e >= E) return;
  const float* x = xF + (size_t)e * 128;
  float s = 0.0f;
  for (int c = 0; c < 128; ++c) s += x[c] * W[c];
  F_st[e] += s;
}

// fused: cbf_out = sbf@W_cbf_out; xS = xF[ba]*(cbf_out@W_cbf_S); S += xS@W_out_S
__global__ __launch_bounds__(256) void outS_kernel(const float* __restrict__ sbf,
    const float* __restrict__ xF, const int* __restrict__ ba,
    const float* __restrict__ Wco, const float* __restrict__ Wcs,
    const float* __restrict__ Wos, float* __restrict__ S_st, int T) {
  __shared__ float wcs[2048];
  __shared__ float wos[384];
  __shared__ float wco[112];
  int tid = threadIdx.x;
  for (int i = tid; i < 2048; i += 256) wcs[i] = Wcs[i];
  for (int i = tid; i < 384; i += 256)  wos[i] = Wos[i];
  if (tid < 112) wco[tid] = Wco[tid];
  __syncthreads();
  int t = blockIdx.x * 256 + tid;
  if (t >= T) return;
  float sb[7];
  #pragma unroll
  for (int l = 0; l < 7; ++l) sb[l] = sbf[(size_t)t*7 + l];
  float cb[16];
  #pragma unroll
  for (int k = 0; k < 16; ++k) {
    float s = 0.0f;
    #pragma unroll
    for (int l = 0; l < 7; ++l) s += sb[l] * wco[l*16 + k];
    cb[k] = s;
  }
  const float* x = xF + (size_t)ba[t] * 128;
  float s0 = 0.0f, s1 = 0.0f, s2 = 0.0f;
  for (int d = 0; d < 128; ++d) {
    float g = 0.0f;
    #pragma unroll
    for (int k = 0; k < 16; ++k) g += cb[k] * wcs[k*128 + d];
    float v = g * x[d];
    s0 += v * wos[d*3]; s1 += v * wos[d*3+1]; s2 += v * wos[d*3+2];
  }
  S_st[(size_t)t*3]   += s0;
  S_st[(size_t)t*3+1] += s1;
  S_st[(size_t)t*3+2] += s2;
}

__global__ void force_scatter_kernel(const float* __restrict__ F_st, const float* __restrict__ V,
                                     const int* __restrict__ idx_t, float* __restrict__ Ft, int E) {
  int e = blockIdx.x * 256 + threadIdx.x;
  if (e >= E) return;
  float f = F_st[e];
  int n = idx_t[e];
  atomicAdd(&Ft[n*3+0], f * V[e*3+0]);
  atomicAdd(&Ft[n*3+1], f * V[e*3+1]);
  atomicAdd(&Ft[n*3+2], f * V[e*3+2]);
}

__global__ void cellinv_kernel(const float* __restrict__ cell, float* __restrict__ cinv, int C) {
  int c = blockIdx.x * 256 + threadIdx.x;
  if (c >= C) return;
  const float* a = cell + c*9;
  float det = a[0]*(a[4]*a[8]-a[5]*a[7]) - a[1]*(a[3]*a[8]-a[5]*a[6]) + a[2]*(a[3]*a[7]-a[4]*a[6]);
  float id = 1.0f / det;
  float* o = cinv + c*9;
  o[0] =  (a[4]*a[8]-a[5]*a[7])*id; o[1] = -(a[1]*a[8]-a[2]*a[7])*id; o[2] =  (a[1]*a[5]-a[2]*a[4])*id;
  o[3] = -(a[3]*a[8]-a[5]*a[6])*id; o[4] =  (a[0]*a[8]-a[2]*a[6])*id; o[5] = -(a[0]*a[5]-a[2]*a[3])*id;
  o[6] =  (a[3]*a[7]-a[4]*a[6])*id; o[7] = -(a[0]*a[7]-a[1]*a[6])*id; o[8] =  (a[0]*a[4]-a[1]*a[3])*id;
}

__global__ void atoms_final_kernel(const float* __restrict__ x, const float* __restrict__ Ft,
                                   const float* __restrict__ cinv, const int* __restrict__ batch_atom,
                                   float* __restrict__ out, int NA) {
  int n = blockIdx.x * 256 + threadIdx.x;
  if (n >= NA) return;
  const float* ci = cinv + batch_atom[n]*9;
  float f0 = Ft[n*3], f1 = Ft[n*3+1], f2 = Ft[n*3+2];
  float g0 = ci[0]*f0 + ci[1]*f1 + ci[2]*f2;
  float g1 = ci[3]*f0 + ci[4]*f1 + ci[5]*f2;
  float g2 = ci[6]*f0 + ci[7]*f1 + ci[8]*f2;
  float x0 = x[n*3] + g0,  x1 = x[n*3+1] + g1, x2 = x[n*3+2] + g2;
  x0 -= floorf(x0); x1 -= floorf(x1); x2 -= floorf(x2);
  out[n*3] = x0; out[n*3+1] = x1; out[n*3+2] = x2;
  float* fo = out + 12288;
  fo[n*3] = g0; fo[n*3+1] = g1; fo[n*3+2] = g2;
}

__global__ void stress_kernel(const float* __restrict__ V, const float* __restrict__ S_st,
                              const int* __restrict__ ba, const int* __restrict__ ca,
                              const int* __restrict__ batch_edge,
                              float* __restrict__ num, float* __restrict__ den, int T) {
  int t = blockIdx.x * 256 + threadIdx.x;
  if (t >= T) return;
  int eb = ba[t], ec = ca[t];
  int b = batch_edge[eb];
  // u_ij = -V[eb], u_ik = -V[ec]; signs cancel in bilinear fields
  float ua[3] = {V[eb*3], V[eb*3+1], V[eb*3+2]};
  float ub[3] = {V[ec*3], V[ec*3+1], V[ec*3+2]};
  float s0 = S_st[(size_t)t*3], s1 = S_st[(size_t)t*3+1], s2 = S_st[(size_t)t*3+2];
  float ab = ua[0]*ub[0] + ua[1]*ub[1] + ua[2]*ub[2];
  float n3inv = rsqrtf(0.5f * (1.0f + ab*ab));   // |f1|=|f2|=1
  #pragma unroll
  for (int i = 0; i < 3; ++i) {
    #pragma unroll
    for (int j = 0; j < 3; ++j) {
      float f1 = ua[i]*ua[j];
      float f2 = ub[i]*ub[j];
      float f3 = 0.5f * (ua[i]*ub[j] + ub[i]*ua[j]) * n3inv;
      atomicAdd(&num[b*9 + i*3 + j], s0*f1 + s1*f2 + s2*f3);
    }
  }
  atomicAdd(&den[b], 1.0f);
}

__global__ void cellfinal_kernel(const float* __restrict__ cell, const float* __restrict__ num,
                                 const float* __restrict__ den, float* __restrict__ out_cellp,
                                 float* __restrict__ out_St, int C) {
  int c = blockIdx.x * 256 + threadIdx.x;
  if (c >= C) return;
  float d = fmaxf(den[c], 1.0f);
  float S[9];
  #pragma unroll
  for (int i = 0; i < 9; ++i) S[i] = num[c*9 + i] / d;
  S[0] += 1.0f; S[4] += 1.0f; S[8] += 1.0f;
  const float* cl = cell + c*9;
  #pragma unroll
  for (int i = 0; i < 3; ++i)
    #pragma unroll
    for (int j = 0; j < 3; ++j) {
      float s = 0.0f;
      #pragma unroll
      for (int k = 0; k < 3; ++k) s += S[i*3+k] * cl[k*3+j];
      out_cellp[c*9 + i*3 + j] = s;
      out_St[c*9 + i*3 + j]    = S[i*3+j];
    }
}

// ---------------------------------------------------------------------------
// Host side
// ---------------------------------------------------------------------------
enum { MODE_NONE = 0, MODE_ACT = 1, MODE_RES = 2, MODE_ACT_GATE = 3, MODE_GATE = 4 };

static inline void gemm(hipStream_t st, const bf16* X, int ldx, const bf16* Wt,
                        float* Yf, bf16* Yb, int ldy, const float* Res, const float* Gate,
                        int M, int K, int N, int mode) {
  dim3 grid(M / 128), block(256);
  switch (N >> 4) {
    case 1: wmma_gemm_kernel<1><<<grid, block, 0, st>>>(X, ldx, Wt, Yf, Yb, ldy, Res, Gate, K, mode); break;
    case 2: wmma_gemm_kernel<2><<<grid, block, 0, st>>>(X, ldx, Wt, Yf, Yb, ldy, Res, Gate, K, mode); break;
    case 4: wmma_gemm_kernel<4><<<grid, block, 0, st>>>(X, ldx, Wt, Yf, Yb, ldy, Res, Gate, K, mode); break;
    default: wmma_gemm_kernel<8><<<grid, block, 0, st>>>(X, ldx, Wt, Yf, Yb, ldy, Res, Gate, K, mode); break;
  }
}

#define GRID1(n) dim3((unsigned)(((n) + 255) / 256)), dim3(256)

// Param pytree offsets (JAX sorted-key flatten order)
static const int OFF_W_cbf3    = 0;
static const int OFF_W_cbf_out = 14336;
static const int OFF_W_edge    = 14448;
static const int OFF_W_rbf3    = 63600;
static const int OFF_W_rbf_h   = 65648;
static const int OFF_W_rbf_out = 67696;
static const int OFF_atom_emb  = 69744;
static const int OFF_INT0      = 82544;
static const int INT_SZ        = 385024;
static const int OFF_OUT0      = 1237616;
static const int OUT_SZ        = 119296;
static const int PARAM_TOTAL   = 1714800;
// within int block
static const int I_W_au0 = 0, I_W_au_rbf = 16384, I_W_ba = 18432, I_W_bil = 34816,
                 I_W_ca = 67584, I_W_concat = 83968, I_W_down = 133120, I_W_trip = 137216,
                 I_W_up_ac = 139264, I_W_up_ca = 147456, I_res_as = 155648, I_res_au = 221184,
                 I_res_bs = 319488, I_res_m = 352256;
// within out block
static const int O_W_F0 = 0, O_W_cbf_S = 16384, O_W_out_F = 18432, O_W_out_S = 18560,
                 O_W_rbf_F = 18944, O_res_F = 20992;

extern "C" void kernel_launch(void* const* d_in, const int* in_sizes, int n_in,
                              void* d_out, int out_size, void* d_ws, size_t ws_size,
                              hipStream_t stream) {
  (void)in_sizes; (void)n_in; (void)out_size; (void)ws_size;
  const int NA = 4096, E = 65536, T = 983040, C = 128;

  const float* cell  = (const float*)d_in[0];
  const float* x     = (const float*)d_in[1];
  const float* e_ij  = (const float*)d_in[2];
  const float* P     = (const float*)d_in[3];
  const int* z       = (const int*)d_in[4];
  const int* idx_s   = (const int*)d_in[5];
  const int* idx_t   = (const int*)d_in[6];
  const int* id_swap = (const int*)d_in[7];
  const int* id3_ba  = (const int*)d_in[8];
  const int* id3_ca  = (const int*)d_in[9];
  const int* batch_edge = (const int*)d_in[10];
  const int* batch_atom = (const int*)d_in[11];
  float* out = (float*)d_out;

  // workspace carve-up (deterministic)
  size_t off = 0;
  char* base = (char*)d_ws;
  auto alloc = [&](size_t bytes) -> void* {
    void* p = base + off;
    off += (bytes + 255) & ~(size_t)255;
    return p;
  };
  bf16*  WT      = (bf16*) alloc((size_t)PARAM_TOTAL * 2);
  float* Vdir    = (float*)alloc((size_t)E * 3 * 4);
  float* Dlen    = (float*)alloc((size_t)E * 4);
  float* rbf_f   = (float*)alloc((size_t)E * 128 * 4);
  bf16*  rbf_b   = (bf16*) alloc((size_t)E * 128 * 2);
  float* sbf     = (float*)alloc((size_t)T * 7 * 4);
  float* h_f     = (float*)alloc((size_t)NA * 128 * 4);
  bf16*  h_b     = (bf16*) alloc((size_t)NA * 128 * 2);
  bf16*  cat_b   = (bf16*) alloc((size_t)E * 384 * 2);
  float* m_f     = (float*)alloc((size_t)E * 128 * 4);
  bf16*  m_b     = (bf16*) alloc((size_t)E * 128 * 2);
  bf16*  rbf3_b  = (bf16*) alloc((size_t)E * 16 * 2);
  bf16*  rbfh_b  = (bf16*) alloc((size_t)E * 16 * 2);
  bf16*  rbfo_b  = (bf16*) alloc((size_t)E * 16 * 2);
  float* rbfW1_f = (float*)alloc((size_t)E * 112 * 4);
  float* gate_f  = (float*)alloc((size_t)E * 128 * 4);
  bf16*  t1_b    = (bf16*) alloc((size_t)E * 128 * 2);
  bf16*  t2_b    = (bf16*) alloc((size_t)E * 128 * 2);
  float* xca_f   = (float*)alloc((size_t)E * 128 * 4);
  float* xba_f   = (float*)alloc((size_t)E * 32 * 4);
  float* acc_f   = (float*)alloc((size_t)E * 224 * 4);
  bf16*  xq_b    = (bf16*) alloc((size_t)E * 512 * 2);
  bf16*  mca_b   = (bf16*) alloc((size_t)E * 64 * 2);
  float* upA_f   = (float*)alloc((size_t)E * 128 * 4);
  float* upB_f   = (float*)alloc((size_t)E * 128 * 4);
  float* xm_f    = (float*)alloc((size_t)E * 128 * 4);
  bf16*  xm_b    = (bf16*) alloc((size_t)E * 128 * 2);
  float* a_f     = (float*)alloc((size_t)NA * 128 * 4);
  bf16*  a_b     = (bf16*) alloc((size_t)NA * 128 * 2);
  float* m2_f    = (float*)alloc((size_t)E * 128 * 4);
  bf16*  m2_b    = (bf16*) alloc((size_t)E * 128 * 2);
  float* xF_f    = (float*)alloc((size_t)E * 128 * 4);
  bf16*  xF_b    = (bf16*) alloc((size_t)E * 128 * 2);
  bf16*  xF0_b   = (bf16*) alloc((size_t)E * 128 * 2);
  float* F_st    = (float*)alloc((size_t)E * 4);
  float* S_st    = (float*)alloc((size_t)T * 3 * 4);
  float* F_t     = (float*)alloc((size_t)NA * 3 * 4);
  float* cinv    = (float*)alloc((size_t)C * 9 * 4);
  float* numS    = (float*)alloc((size_t)C * 9 * 4);
  float* denS    = (float*)alloc((size_t)C * 4);

  // ---- weight conversion (fp32 [K,N] -> bf16 [N,K]) ----
  auto prep = [&](int poff, int K, int N) {
    int n = K * N;
    wprep_kernel<<<GRID1(n), 0, stream>>>(P + poff, WT + poff, K, N);
  };
  prep(OFF_W_edge, 384, 128);
  prep(OFF_W_rbf3, 128, 16);
  prep(OFF_W_rbf_h, 128, 16);
  prep(OFF_W_rbf_out, 128, 16);
  for (int s = 0; s < 7; ++s) prep(OFF_W_cbf3 + s * 2048, 128, 16);
  for (int b = 0; b < 3; ++b) {
    int B = OFF_INT0 + b * INT_SZ;
    prep(B + I_W_ca, 128, 128);   prep(B + I_W_ba, 128, 128);
    prep(B + I_W_trip, 16, 128);  prep(B + I_W_down, 128, 32);
    prep(B + I_W_bil, 512, 64);   prep(B + I_W_up_ca, 64, 128);
    prep(B + I_W_up_ac, 64, 128); prep(B + I_W_au_rbf, 16, 128);
    prep(B + I_W_au0, 128, 128);  prep(B + I_W_concat, 384, 128);
    prep(B + I_res_bs, 128, 128); prep(B + I_res_bs + 16384, 128, 128);
    for (int r = 0; r < 2; ++r) { prep(B + I_res_as + r*32768, 128, 128); prep(B + I_res_as + r*32768 + 16384, 128, 128); }
    for (int r = 0; r < 3; ++r) { prep(B + I_res_au + r*32768, 128, 128); prep(B + I_res_au + r*32768 + 16384, 128, 128); }
    prep(B + I_res_m, 128, 128);  prep(B + I_res_m + 16384, 128, 128);
  }
  for (int b = 0; b < 4; ++b) {
    int B = OFF_OUT0 + b * OUT_SZ;
    prep(B + O_W_rbf_F, 16, 128); prep(B + O_W_F0, 128, 128);
    for (int r = 0; r < 3; ++r) { prep(B + O_res_F + r*32768, 128, 128); prep(B + O_res_F + r*32768 + 16384, 128, 128); }
  }

  // ---- geometry / bases ----
  (void)hipMemsetAsync(F_st, 0, (size_t)E * 4, stream);
  (void)hipMemsetAsync(S_st, 0, (size_t)T * 3 * 4, stream);
  (void)hipMemsetAsync(F_t, 0, (size_t)NA * 3 * 4, stream);
  (void)hipMemsetAsync(numS, 0, (size_t)C * 9 * 4, stream);
  (void)hipMemsetAsync(denS, 0, (size_t)C * 4, stream);

  edge_vec_kernel<<<GRID1(E), 0, stream>>>(cell, e_ij, batch_edge, Vdir, Dlen, E);
  rbf_kernel<<<GRID1(E * 128), 0, stream>>>(Dlen, rbf_f, rbf_b, E);
  sbf_kernel<<<GRID1(T), 0, stream>>>(Vdir, id3_ba, id3_ca, sbf, T);
  embed_kernel<<<GRID1(NA * 128), 0, stream>>>(P + OFF_atom_emb, z, h_f, h_b, NA);
  concat_kernel<<<GRID1(E * 384), 0, stream>>>(h_b, idx_s, idx_t, rbf_b, cat_b, E);
  gemm(stream, cat_b, 384, WT + OFF_W_edge, m_f, m_b, 128, nullptr, nullptr, E, 384, 128, MODE_ACT);

  gemm(stream, rbf_b, 128, WT + OFF_W_rbf3, nullptr, rbf3_b, 16, nullptr, nullptr, E, 128, 16, MODE_NONE);
  for (int s = 0; s < 7; ++s)
    gemm(stream, rbf_b, 128, WT + OFF_W_cbf3 + s * 2048, rbfW1_f + s * 16, nullptr, 112,
         nullptr, nullptr, E, 128, 16, MODE_NONE);
  gemm(stream, rbf_b, 128, WT + OFF_W_rbf_h, nullptr, rbfh_b, 16, nullptr, nullptr, E, 128, 16, MODE_NONE);
  gemm(stream, rbf_b, 128, WT + OFF_W_rbf_out, nullptr, rbfo_b, 16, nullptr, nullptr, E, 128, 16, MODE_NONE);

  auto res_layer_run = [&](float* xf, bf16* xb, int M, int W1off, int W2off) {
    gemm(stream, xb, 128, WT + W1off, nullptr, t1_b, 128, nullptr, nullptr, M, 128, 128, MODE_ACT);
    gemm(stream, t1_b, 128, WT + W2off, xf, xb, 128, xf, nullptr, M, 128, 128, MODE_RES);
  };

  auto out_block = [&](int B) {
    gemm(stream, rbfo_b, 16, WT + B + O_W_rbf_F, gate_f, nullptr, 128, nullptr, nullptr, E, 16, 128, MODE_NONE);
    ew_mul_tobf<<<GRID1(E * 128), 0, stream>>>(m_f, gate_f, xF0_b, E * 128);
    gemm(stream, xF0_b, 128, WT + B + O_W_F0, xF_f, xF_b, 128, nullptr, nullptr, E, 128, 128, MODE_ACT);
    for (int r = 0; r < 3; ++r)
      res_layer_run(xF_f, xF_b, E, B + O_res_F + r*32768, B + O_res_F + r*32768 + 16384);
    outF_kernel<<<GRID1(E), 0, stream>>>(xF_f, P + B + O_W_out_F, F_st, E);
    outS_kernel<<<GRID1(T), 0, stream>>>(sbf, xF_f, id3_ba, P + OFF_W_cbf_out,
                                         P + B + O_W_cbf_S, P + B + O_W_out_S, S_st, T);
  };

  auto int_block = [&](int B) {
    gemm(stream, m_b, 128, WT + B + I_W_ca, xca_f, nullptr, 128, nullptr, nullptr, E, 128, 128, MODE_ACT);
    gemm(stream, rbf3_b, 16, WT + B + I_W_trip, gate_f, nullptr, 128, nullptr, nullptr, E, 16, 128, MODE_NONE);
    gemm(stream, m_b, 128, WT + B + I_W_ba, nullptr, t2_b, 128, nullptr, gate_f, E, 128, 128, MODE_ACT_GATE);
    gemm(stream, t2_b, 128, WT + B + I_W_down, xba_f, nullptr, 32, nullptr, nullptr, E, 128, 32, MODE_ACT);
    trip_acc_kernel<<<GRID1(E * 32), 0, stream>>>(sbf, xba_f, acc_f, E);
    xq_kernel<<<dim3(E), dim3(256), 0, stream>>>(rbfW1_f, acc_f, xq_b);
    gemm(stream, xq_b, 512, WT + B + I_W_bil, nullptr, mca_b, 64, nullptr, nullptr, E, 512, 64, MODE_NONE);
    gemm(stream, mca_b, 64, WT + B + I_W_up_ca, upA_f, nullptr, 128, nullptr, nullptr, E, 64, 128, MODE_ACT);
    gemm(stream, mca_b, 64, WT + B + I_W_up_ac, upB_f, nullptr, 128, nullptr, nullptr, E, 64, 128, MODE_ACT);
    ew_combine<<<GRID1(E * 128), 0, stream>>>(xca_f, upA_f, upB_f, id_swap, xm_f, xm_b, E);
    res_layer_run(xm_f, xm_b, E, B + I_res_bs, B + I_res_bs + 16384);
    ew_addscale<<<GRID1(E * 128), 0, stream>>>(xm_f, m_f, xm_f, xm_b, E * 128);
    for (int r = 0; r < 2; ++r)
      res_layer_run(xm_f, xm_b, E, B + I_res_as + r*32768, B + I_res_as + r*32768 + 16384);
    // atom update
    gemm(stream, rbfh_b, 16, WT + B + I_W_au_rbf, gate_f, nullptr, 128, nullptr, nullptr, E, 16, 128, MODE_NONE);
    (void)hipMemsetAsync(a_f, 0, (size_t)NA * 128 * 4, stream);
    scatter_atom_kernel<<<GRID1(E * 128), 0, stream>>>(xm_f, gate_f, idx_t, a_f, E);
    f2b_kernel<<<GRID1(NA * 128), 0, stream>>>(a_f, a_b, NA * 128);
    gemm(stream, a_b, 128, WT + B + I_W_au0, a_f, a_b, 128, nullptr, nullptr, NA, 128, 128, MODE_ACT);
    for (int r = 0; r < 3; ++r)
      res_layer_run(a_f, a_b, NA, B + I_res_au + r*32768, B + I_res_au + r*32768 + 16384);
    ew_addscale<<<GRID1(NA * 128), 0, stream>>>(h_f, a_f, h_f, h_b, NA * 128);
    // edge update
    concat_kernel<<<GRID1(E * 384), 0, stream>>>(h_b, idx_s, idx_t, xm_b, cat_b, E);
    gemm(stream, cat_b, 384, WT + B + I_W_concat, m2_f, m2_b, 128, nullptr, nullptr, E, 384, 128, MODE_ACT);
    res_layer_run(m2_f, m2_b, E, B + I_res_m, B + I_res_m + 16384);
    ew_addscale<<<GRID1(E * 128), 0, stream>>>(m_f, m2_f, m_f, m_b, E * 128);
  };

  out_block(OFF_OUT0);
  for (int i = 0; i < 3; ++i) {
    int_block(OFF_INT0 + i * INT_SZ);
    out_block(OFF_OUT0 + (i + 1) * OUT_SZ);
  }

  // ---- forces / positions / stress / cell ----
  force_scatter_kernel<<<GRID1(E), 0, stream>>>(F_st, Vdir, idx_t, F_t, E);
  cellinv_kernel<<<GRID1(C), 0, stream>>>(cell, cinv, C);
  atoms_final_kernel<<<GRID1(NA), 0, stream>>>(x, F_t, cinv, batch_atom, out, NA);
  (void)hipMemcpyAsync(out + 24576, h_f, (size_t)NA * 128 * 4, hipMemcpyDeviceToDevice, stream);
  stress_kernel<<<GRID1(T), 0, stream>>>(Vdir, S_st, id3_ba, id3_ca, batch_edge, numS, denS, T);
  cellfinal_kernel<<<GRID1(C), 0, stream>>>(cell, numS, denS, out + 548864, out + 550016, C);
}
